// Sampler_18657337934661
// MI455X (gfx1250) — matrix operand order repaired and checked by reading, not compiled, not run on hardware
//
#include <hip/hip_runtime.h>

// Problem constants (from reference)
#define NWAY   64
#define KSHOT  5
#define DIM    1024
#define NQ     100000
#define TOPK   5
#define MROWS  (NWAY * KSHOT)          // 320
#define MTILES (MROWS / 16)            // 20
#define QWG    64                      // queries per workgroup (4 waves x 16)
#define NUMWG  ((NQ + QWG - 1) / QWG)  // 1563

typedef __attribute__((ext_vector_type(16))) _Float16 v16h;
typedef __attribute__((ext_vector_type(8)))  _Float16 v8h;
typedef __attribute__((ext_vector_type(8)))  float    v8f;

// ---------------------------------------------------------------------------
// Kernel 1: normalize prototypes, store as f16 panel [MROWS][DIM].
// ---------------------------------------------------------------------------
__global__ __launch_bounds__(256) void pnorm_kernel(
    const float* __restrict__ p, _Float16* __restrict__ ph) {
  __shared__ float red[8];
  int row = blockIdx.x;  // 0..319
  const float* src = p + (size_t)row * DIM;
  float s = 0.f;
  for (int k = threadIdx.x; k < DIM; k += 256) {
    float v = src[k];
    s += v * v;
  }
  for (int off = 16; off; off >>= 1) s += __shfl_xor(s, off, 32);
  if ((threadIdx.x & 31) == 0) red[threadIdx.x >> 5] = s;
  __syncthreads();
  float tot = 0.f;
#pragma unroll
  for (int i = 0; i < 8; ++i) tot += red[i];
  float inv = rsqrtf(tot);
  for (int k = threadIdx.x; k < DIM; k += 256)
    ph[(size_t)row * DIM + k] = (_Float16)(src[k] * inv);
}

// ---------------------------------------------------------------------------
// Kernel 2: fused similarity GEMM (f16 WMMA, f32 accum) with inline query
// L2-norm computation + per-block top-5.
// 4 waves x 16 queries; each wave covers all 20 M-tiles for its queries.
// Queries are streamed from HBM exactly once.
// ---------------------------------------------------------------------------
__global__ __launch_bounds__(128) void sim_topk_kernel(
    const float* __restrict__ queries,
    const _Float16* __restrict__ ph,
    float* __restrict__ candVal,
    int* __restrict__ candIdx) {
  const int tid  = threadIdx.x;
  const int wave = tid >> 5;
  const int lane = tid & 31;
  const int hi   = lane >> 4;   // lane-group (0/1)
  const int nl   = lane & 15;

  const int q0   = blockIdx.x * QWG + wave * 16;
  const int qcol = q0 + nl;
  const int qc   = qcol < NQ ? qcol : (NQ - 1);
  const bool valid = qcol < NQ;

  // B layout (32x16 f16): n = lane&15, K base = hi*8, halves 0..7 -> K+0..7,
  // halves 8..15 -> K+16..23 (same as A).
  const float* bBase = queries + (size_t)qc * DIM + hi * 8;
  // A layout (16x32 f16): m = lane&15, K base = hi*8, halves 0..7 -> K+0..7,
  // halves 8..15 -> K+16..23.
  const _Float16* aBase = ph + (size_t)nl * DIM + hi * 8;

  v8f zero = {0.f, 0.f, 0.f, 0.f, 0.f, 0.f, 0.f, 0.f};
  v8f acc[MTILES];
#pragma unroll
  for (int t = 0; t < MTILES; ++t) acc[t] = zero;

  float sq = 0.f;  // partial sum of squares of this lane's query K-slices

  for (int k0 = 0; k0 < DIM; k0 += 32) {
    // Prefetch next K-chunk of this wave's query rows.
    __builtin_prefetch(bBase + k0 + 32, 0, 1);

    // Build B fragment: two 8-f32 runs (k+0..7, k+16..23) -> 16 halves; accumulate sumsq.
    v16h b;
    {
      const float* bp = bBase + k0;
      float4 f0 = *(const float4*)(bp + 0);
      float4 f1 = *(const float4*)(bp + 4);
      float4 f2 = *(const float4*)(bp + 16);
      float4 f3 = *(const float4*)(bp + 20);
      sq += f0.x * f0.x + f0.y * f0.y + f0.z * f0.z + f0.w * f0.w;
      sq += f1.x * f1.x + f1.y * f1.y + f1.z * f1.z + f1.w * f1.w;
      sq += f2.x * f2.x + f2.y * f2.y + f2.z * f2.z + f2.w * f2.w;
      sq += f3.x * f3.x + f3.y * f3.y + f3.z * f3.z + f3.w * f3.w;
      b[0]  = (_Float16)f0.x; b[1]  = (_Float16)f0.y;
      b[2]  = (_Float16)f0.z; b[3]  = (_Float16)f0.w;
      b[4]  = (_Float16)f1.x; b[5]  = (_Float16)f1.y;
      b[6]  = (_Float16)f1.z; b[7]  = (_Float16)f1.w;
      b[8]  = (_Float16)f2.x; b[9]  = (_Float16)f2.y;
      b[10] = (_Float16)f2.z; b[11] = (_Float16)f2.w;
      b[12] = (_Float16)f3.x; b[13] = (_Float16)f3.y;
      b[14] = (_Float16)f3.z; b[15] = (_Float16)f3.w;
    }
#pragma unroll
    for (int t = 0; t < MTILES; ++t) {
      const _Float16* ap = aBase + (size_t)(t * 16) * DIM + k0;
      v8h alo = *(const v8h*)(ap);
      v8h ahi = *(const v8h*)(ap + 16);
      v16h a;
#pragma unroll
      for (int i = 0; i < 8; ++i) { a[i] = alo[i]; a[i + 8] = ahi[i]; }
      acc[t] = __builtin_amdgcn_wmma_f32_16x16x32_f16(
          false, a, false, b, (short)0, acc[t], false, false);
    }
  }

  // Lane (n, hi) covered k%32 in [8hi,8hi+8) and [16+8hi,24+8hi):
  // pair them to get the full ||q||^2 in both lanes, then derive 1/||q||.
  sq += __shfl_xor(sq, 16, 32);
  const float scale = rsqrtf(sq);

  // Stage each 16-row tile to LDS and pick the block-local top-5 per row.
  __shared__ float sval[16][QWG];
  const int col = wave * 16 + nl;

#pragma unroll
  for (int t = 0; t < MTILES; ++t) {
    __syncthreads();
#pragma unroll
    for (int r = 0; r < 8; ++r) {
      int rl = r + 8 * hi;                 // row within tile (C layout)
      float v = acc[t][r] * scale;
      sval[rl][col] = valid ? v : -1e30f;
    }
    __syncthreads();
    if (tid < 16) {
      float bv[TOPK];
      int bc[TOPK];
#pragma unroll
      for (int j = 0; j < TOPK; ++j) { bv[j] = -1e30f; bc[j] = 0; }
      for (int c = 0; c < QWG; ++c) {
        float v = sval[tid][c];
        if (v > bv[TOPK - 1]) {
          int pos = TOPK - 1;
          while (pos > 0 && v > bv[pos - 1]) {
            bv[pos] = bv[pos - 1];
            bc[pos] = bc[pos - 1];
            --pos;
          }
          bv[pos] = v;
          bc[pos] = c;
        }
      }
      int rowg = t * 16 + tid;
      size_t base = ((size_t)rowg * NUMWG + blockIdx.x) * TOPK;
#pragma unroll
      for (int j = 0; j < TOPK; ++j) {
        candVal[base + j] = bv[j];
        candIdx[base + j] = blockIdx.x * QWG + bc[j];
      }
    }
  }
}

// ---------------------------------------------------------------------------
// Kernel 3: per-row reduction of block candidates -> global sorted top-5 idx.
// ---------------------------------------------------------------------------
__global__ __launch_bounds__(256) void topk_reduce_kernel(
    const float* __restrict__ candVal,
    const int* __restrict__ candIdx,
    int* __restrict__ finalIdx) {
  __shared__ float sv[256 * TOPK];
  __shared__ int   si[256 * TOPK];
  const int row = blockIdx.x;
  const float* cv = candVal + (size_t)row * NUMWG * TOPK;
  const int*   ci = candIdx + (size_t)row * NUMWG * TOPK;
  const int total = NUMWG * TOPK;

  float bv[TOPK];
  int   bi[TOPK];
#pragma unroll
  for (int j = 0; j < TOPK; ++j) { bv[j] = -1e30f; bi[j] = 0; }
  for (int i = threadIdx.x; i < total; i += 256) {
    float v = cv[i];
    if (v > bv[TOPK - 1]) {
      int idx = ci[i];
      int pos = TOPK - 1;
      while (pos > 0 && v > bv[pos - 1]) {
        bv[pos] = bv[pos - 1];
        bi[pos] = bi[pos - 1];
        --pos;
      }
      bv[pos] = v;
      bi[pos] = idx;
    }
  }
#pragma unroll
  for (int j = 0; j < TOPK; ++j) {
    sv[threadIdx.x * TOPK + j] = bv[j];
    si[threadIdx.x * TOPK + j] = bi[j];
  }
  __syncthreads();
  if (threadIdx.x == 0) {
    float fv[TOPK];
    int   fi[TOPK];
#pragma unroll
    for (int j = 0; j < TOPK; ++j) { fv[j] = -1e30f; fi[j] = 0; }
    for (int i = 0; i < 256 * TOPK; ++i) {
      float v = sv[i];
      if (v > fv[TOPK - 1]) {
        int idx = si[i];
        int pos = TOPK - 1;
        while (pos > 0 && v > fv[pos - 1]) {
          fv[pos] = fv[pos - 1];
          fi[pos] = fi[pos - 1];
          --pos;
        }
        fv[pos] = v;
        fi[pos] = idx;
      }
    }
#pragma unroll
    for (int j = 0; j < TOPK; ++j) finalIdx[row * TOPK + j] = fi[j];
  }
}

// ---------------------------------------------------------------------------
// Kernel 4: gather raw query rows into the output.
// ---------------------------------------------------------------------------
__global__ __launch_bounds__(256) void gather_kernel(
    const float* __restrict__ queries,
    const int* __restrict__ finalIdx,
    float* __restrict__ out) {
  int b = blockIdx.x;  // 0 .. MROWS*TOPK-1; out offset == b*DIM (row-major)
  int src = finalIdx[b];
  const float4* s = (const float4*)(queries + (size_t)src * DIM);
  float4* d = (float4*)(out + (size_t)b * DIM);
  for (int i = threadIdx.x; i < DIM / 4; i += 256) d[i] = s[i];
}

// ---------------------------------------------------------------------------
extern "C" void kernel_launch(void* const* d_in, const int* in_sizes, int n_in,
                              void* d_out, int out_size, void* d_ws,
                              size_t ws_size, hipStream_t stream) {
  (void)in_sizes; (void)n_in; (void)out_size; (void)ws_size;
  const float* prototypes = (const float*)d_in[0];  // [320][1024] f32
  const float* queries    = (const float*)d_in[1];  // [100000][1024] f32
  float* out = (float*)d_out;                       // [320*5][1024] f32

  // Workspace carve-up (256B aligned).
  char* ws = (char*)d_ws;
  size_t off = 0;
  auto carve = [&](size_t bytes) {
    size_t o = off;
    off = (off + bytes + 255) & ~(size_t)255;
    return o;
  };
  _Float16* ph       = (_Float16*)(ws + carve((size_t)MROWS * DIM * sizeof(_Float16)));
  float*    candVal  = (float*)   (ws + carve((size_t)MROWS * NUMWG * TOPK * sizeof(float)));
  int*      candIdx  = (int*)     (ws + carve((size_t)MROWS * NUMWG * TOPK * sizeof(int)));
  int*      finalIdx = (int*)     (ws + carve((size_t)MROWS * TOPK * sizeof(int)));

  pnorm_kernel<<<MROWS, 256, 0, stream>>>(prototypes, ph);
  sim_topk_kernel<<<NUMWG, 128, 0, stream>>>(queries, ph, candVal, candIdx);
  topk_reduce_kernel<<<MROWS, 256, 0, stream>>>(candVal, candIdx, finalIdx);
  gather_kernel<<<MROWS * TOPK, 256, 0, stream>>>(queries, finalIdx, out);
}
